// GAT_67559835566268
// MI455X (gfx1250) — compile-verified, matrix operand order checked by reading, not run on hardware
//
#include <hip/hip_runtime.h>
#include <hip/hip_bf16.h>

typedef float     v2f  __attribute__((ext_vector_type(2)));
typedef float     v4f  __attribute__((ext_vector_type(4)));
typedef float     v8f  __attribute__((ext_vector_type(8)));
typedef _Float16  v8h  __attribute__((ext_vector_type(8)));
typedef _Float16  v16h __attribute__((ext_vector_type(16)));
typedef int       v4i  __attribute__((ext_vector_type(4)));
typedef int       v8i  __attribute__((ext_vector_type(8)));

#define N_NODES 4096
#define NWORDS  128      // 4096 / 32

// ---------------------------------------------------------------------------
// TDM: issue a 2D tensor_load_to_lds (D# built in SGPRs, per CDNA5 ISA).
// data_size fixed to 4 bytes.  Wave-level op: call from one wave only, then
// s_wait_tensorcnt before the workgroup barrier.
// ---------------------------------------------------------------------------
__device__ __forceinline__ void tdm_load_2d_b32(
    const void* gptr, unsigned lds_off,
    unsigned tdim0, unsigned tdim1,
    unsigned tile0, unsigned tile1, unsigned stride0)
{
    unsigned long long ga = (unsigned long long)gptr;
    v4i g0; v8i g1; v4i gz = {0, 0, 0, 0};
    // group0: [1:0]=count=1 | lds_addr | global_addr[56:0] | type=2
    g0[0] = 1;
    g0[1] = (int)lds_off;
    g0[2] = (int)(unsigned)(ga & 0xFFFFFFFFull);
    g0[3] = (int)((unsigned)((ga >> 32) & 0x01FFFFFFu) | 0x80000000u);
    // group1: data_size=2 (4B); dims/tiles/stride; no pad/iterate/multicast
    g1[0] = (int)(2u << 16);
    g1[1] = (int)((tdim0 & 0xFFFFu) << 16);
    g1[2] = (int)((tdim0 >> 16) | ((tdim1 & 0xFFFFu) << 16));
    g1[3] = (int)((tdim1 >> 16) | (tile0 << 16));
    g1[4] = (int)(tile1 & 0xFFFFu);
    g1[5] = (int)stride0;      // tensor_dim0_stride[31:0]
    g1[6] = 0;                 // stride0[47:32]=0, stride1[15:0]=0
    g1[7] = 0;
    asm volatile("tensor_load_to_lds %0, %1, %2, %3"
                 :: "s"(g0), "s"(g1), "s"(gz), "s"(gz)
                 : "memory");
}

// ---------------------------------------------------------------------------
// Kernel 0: pack adj (int32 0/1, 64 MB) into a bitmask (2 MB).  One wave per
// row; lane reads adj[row][w*32+lane] (coalesced 128B), ballot packs 32 bits.
// ---------------------------------------------------------------------------
__global__ __launch_bounds__(256) void pack_adj_kernel(
    const int* __restrict__ adj, unsigned* __restrict__ mask)
{
    int gwave = (blockIdx.x * blockDim.x + threadIdx.x) >> 5;   // row
    int lane  = threadIdx.x & 31;
    if (gwave >= N_NODES) return;
    const int* row = adj + (size_t)gwave * N_NODES;
    for (int w = 0; w < NWORDS; ++w) {
        int v = row[w * 32 + lane];
        unsigned bal = __builtin_amdgcn_ballot_w32(v > 0);
        if (lane == 0) mask[gwave * NWORDS + w] = bal;
    }
}

// ---------------------------------------------------------------------------
// Kernel 1: wh = X @ W + b   (full f32 via V_WMMA_F32_16X16X4_F32)
//   X:[4096,K] f32, W:[nh][K][64], bias:[nh][64]
//   out: wh_f32:[nh][4096][64]  and whT_f16:[nh][64][4096] (B-operand copy)
// One wave computes a 16x64 output tile.  grid = (256, nh), block = 32.
// ---------------------------------------------------------------------------
__global__ __launch_bounds__(32) void gemm_wh_kernel(
    const float* __restrict__ X, const float* __restrict__ W,
    const float* __restrict__ bias,
    float* __restrict__ wh, _Float16* __restrict__ whT, int K)
{
    const int tile = blockIdx.x, head = blockIdx.y;
    const int lane = threadIdx.x;
    const int lm   = lane & 15;          // M for A, N for B/D
    const int hi   = lane >> 4;          // which K half-pair
    const int i0   = tile * 16;
    const float* Wh   = W + (size_t)head * K * 64;
    const float* xrow = X + (size_t)(i0 + lm) * K + hi * 2;

    v8f c0 = {}, c1 = {}, c2 = {}, c3 = {};
    for (int kb = 0; kb < K; kb += 4) {
        v2f a = *(const v2f*)(xrow + kb);
        const float* wp = Wh + (size_t)(kb + hi * 2) * 64 + lm;
        v2f b0, b1, b2, b3;
        b0.x = wp[0];  b0.y = wp[64];
        b1.x = wp[16]; b1.y = wp[64 + 16];
        b2.x = wp[32]; b2.y = wp[64 + 32];
        b3.x = wp[48]; b3.y = wp[64 + 48];
        c0 = __builtin_amdgcn_wmma_f32_16x16x4_f32(false, a, false, b0, (short)0, c0, false, false);
        c1 = __builtin_amdgcn_wmma_f32_16x16x4_f32(false, a, false, b1, (short)0, c1, false, false);
        c2 = __builtin_amdgcn_wmma_f32_16x16x4_f32(false, a, false, b2, (short)0, c2, false, false);
        c3 = __builtin_amdgcn_wmma_f32_16x16x4_f32(false, a, false, b3, (short)0, c3, false, false);
    }
    const float* bp = bias + head * 64;
    _Float16* wT = whT + (size_t)head * 64 * N_NODES;
#pragma unroll
    for (int r = 0; r < 8; ++r) {
        int mm = i0 + r + hi * 8;
        float* wrow = wh + ((size_t)head * N_NODES + mm) * 64;
        float v;
        v = c0[r] + bp[lm];      wrow[lm]      = v; wT[(size_t)(lm)      * N_NODES + mm] = (_Float16)v;
        v = c1[r] + bp[16 + lm]; wrow[16 + lm] = v; wT[(size_t)(16 + lm) * N_NODES + mm] = (_Float16)v;
        v = c2[r] + bp[32 + lm]; wrow[32 + lm] = v; wT[(size_t)(32 + lm) * N_NODES + mm] = (_Float16)v;
        v = c3[r] + bp[48 + lm]; wrow[48 + lm] = v; wT[(size_t)(48 + lm) * N_NODES + mm] = (_Float16)v;
    }
}

// ---------------------------------------------------------------------------
// Kernel 2: rank-1 score vectors  e[i] = wh[i].a_src + bias,  f[j] = wh[j].a_dst
// ---------------------------------------------------------------------------
__global__ __launch_bounds__(256) void ef_kernel(
    const float* __restrict__ wh, const float* __restrict__ a_src,
    const float* __restrict__ a_dst, const float* __restrict__ a_b,
    float* __restrict__ e, float* __restrict__ f, int nh)
{
    int idx = blockIdx.x * blockDim.x + threadIdx.x;
    if (idx >= nh * N_NODES) return;
    int head = idx >> 12;
    const float* w  = wh + (size_t)idx * 64;
    const float* as = a_src + head * 64;
    const float* ad = a_dst + head * 64;
    float se = 0.f, sf = 0.f;
#pragma unroll 8
    for (int d = 0; d < 64; ++d) { float v = w[d]; se += v * as[d]; sf += v * ad[d]; }
    e[idx] = se + a_b[head];
    f[idx] = sf;
}

// ---------------------------------------------------------------------------
// Kernel 3: fused masked attention  out = elu( softmax(mask(lrelu(e_i+f_j))) @ wh )
// grid = (256 row-tiles, nh heads), block = 128 (4 waves).
// f-vector (16KB) + bitmask tile (8KB) staged into LDS by TDM (wave 0,
// tensor_load_to_lds + s_wait_tensorcnt).  Each wave owns a 1024-wide j
// chunk, builds f16 unnormalized-softmax A tiles (ds_load_b128 for f),
// runs V_WMMA_F32_16X16X32_F16 against whT, waves combine via ds_add_f32.
// Stabilizer = unmasked global max of f (valid: leaky-relu is monotone).
// ---------------------------------------------------------------------------
__global__ __launch_bounds__(128) void attn_kernel(
    const unsigned* __restrict__ mask,
    const float* __restrict__ e, const float* __restrict__ f,
    const _Float16* __restrict__ whT,
    float* __restrict__ out, int outStride)
{
    __shared__ float    s_f[N_NODES];
    __shared__ unsigned s_mask[16][NWORDS];
    __shared__ float    s_acc[16][64];
    __shared__ float    s_sum[16];
    __shared__ float    s_red[128];

    const int head = blockIdx.y;
    const int i0   = blockIdx.x * 16;
    const int t    = threadIdx.x;
    const int wave = t >> 5, lane = t & 31;

    const float* fh = f + (size_t)head * N_NODES;

    if (wave == 0) {
        // TDM: f row-vector (1 x 4096 f32) and mask tile (16 x 128 words)
        tdm_load_2d_b32(fh, (unsigned)(size_t)&s_f[0],
                        N_NODES, 1, N_NODES, 1, N_NODES);
        tdm_load_2d_b32(mask + (size_t)i0 * NWORDS, (unsigned)(size_t)&s_mask[0][0],
                        NWORDS, 16, NWORDS, 16, NWORDS);
        asm volatile("s_wait_tensorcnt 0x0" ::: "memory");
    }
    for (int w = t; w < 16 * 64; w += 128) ((float*)s_acc)[w] = 0.f;
    if (t < 16) s_sum[t] = 0.f;
    __syncthreads();

    // global (unmasked) max of f -> softmax stabilizer
    float fm = -3.0e38f;
    for (int j = t; j < N_NODES; j += 128) fm = fmaxf(fm, s_f[j]);
    s_red[t] = fm; __syncthreads();
    for (int off = 64; off > 0; off >>= 1) {
        if (t < off) s_red[t] = fmaxf(s_red[t], s_red[t + off]);
        __syncthreads();
    }
    const float Fmax = s_red[0];

    const int lm = lane & 15;         // row M for A, col N for B/D
    const int hi = lane >> 4;
    const int r0 = hi * 8;            // K-run start within 32-wide step

    const float em = e[(size_t)head * N_NODES + i0 + lm];
    float sm = em + Fmax;
    const float rowmax = sm > 0.f ? sm : 0.2f * sm;

    const _Float16* wT = whT + (size_t)head * 64 * N_NODES;
    v8f   cacc[4] = {v8f{}, v8f{}, v8f{}, v8f{}};
    float psum = 0.f;

    const int jlo = wave * 1024, jhi = jlo + 1024;
#pragma unroll 1
    for (int jb = jlo; jb < jhi; jb += 32) {
        unsigned word = s_mask[lm][jb >> 5];
        // f values for this lane's two contiguous K-runs: 4x ds_load_b128
        v4f f0 = *(const v4f*)&s_f[jb + r0];
        v4f f1 = *(const v4f*)&s_f[jb + r0 + 4];
        v4f f2 = *(const v4f*)&s_f[jb + 16 + r0];
        v4f f3 = *(const v4f*)&s_f[jb + 16 + r0 + 4];
        v16h a;
#pragma unroll
        for (int kk = 0; kk < 4; ++kk) {
            float x, p;
            x = em + f0[kk]; x = x > 0.f ? x : 0.2f * x;
            p = ((word >> (r0 + kk)) & 1u) ? __expf(x - rowmax) : 0.f;
            psum += p; a[kk] = (_Float16)p;
            x = em + f1[kk]; x = x > 0.f ? x : 0.2f * x;
            p = ((word >> (r0 + 4 + kk)) & 1u) ? __expf(x - rowmax) : 0.f;
            psum += p; a[4 + kk] = (_Float16)p;
            x = em + f2[kk]; x = x > 0.f ? x : 0.2f * x;
            p = ((word >> (16 + r0 + kk)) & 1u) ? __expf(x - rowmax) : 0.f;
            psum += p; a[8 + kk] = (_Float16)p;
            x = em + f3[kk]; x = x > 0.f ? x : 0.2f * x;
            p = ((word >> (16 + r0 + 4 + kk)) & 1u) ? __expf(x - rowmax) : 0.f;
            psum += p; a[12 + kk] = (_Float16)p;
        }
#pragma unroll
        for (int tt = 0; tt < 4; ++tt) {
            const _Float16* col = wT + (size_t)(tt * 16 + lm) * N_NODES + jb;
            v8h blo = *(const v8h*)(col + r0);
            v8h bhi = *(const v8h*)(col + 16 + r0);
            v16h b = __builtin_shufflevector(blo, bhi,
                0, 1, 2, 3, 4, 5, 6, 7, 8, 9, 10, 11, 12, 13, 14, 15);
            cacc[tt] = __builtin_amdgcn_wmma_f32_16x16x32_f16(
                false, a, false, b, (short)0, cacc[tt], false, false);
        }
    }

    atomicAdd(&s_sum[lm], psum);
#pragma unroll
    for (int r = 0; r < 8; ++r) {
        int mm = r + hi * 8;
        atomicAdd(&s_acc[mm][lm],      cacc[0][r]);
        atomicAdd(&s_acc[mm][16 + lm], cacc[1][r]);
        atomicAdd(&s_acc[mm][32 + lm], cacc[2][r]);
        atomicAdd(&s_acc[mm][48 + lm], cacc[3][r]);
    }
    __syncthreads();

    for (int w = t; w < 16 * 64; w += 128) {
        int mm = w >> 6, d = w & 63;
        float v = s_acc[mm][d] / (s_sum[mm] + 1e-30f);
        v = v > 0.f ? v : (__expf(v) - 1.f);     // ELU
        out[(size_t)(i0 + mm) * outStride + head * 64 + d] = v;
    }
}

// ---------------------------------------------------------------------------
extern "C" void kernel_launch(void* const* d_in, const int* in_sizes, int n_in,
                              void* d_out, int out_size, void* d_ws, size_t ws_size,
                              hipStream_t stream)
{
    const float* h      = (const float*)d_in[0];
    const int*   adj    = (const int*)  d_in[1];
    const float* Wh     = (const float*)d_in[2];
    const float* bWh    = (const float*)d_in[3];
    const float* a_src  = (const float*)d_in[4];
    const float* a_dst  = (const float*)d_in[5];
    const float* a_b    = (const float*)d_in[6];
    const float* Wo     = (const float*)d_in[7];
    const float* bWo    = (const float*)d_in[8];
    const float* ao_src = (const float*)d_in[9];
    const float* ao_dst = (const float*)d_in[10];
    const float* ao_b   = (const float*)d_in[11];

    char* ws = (char*)d_ws;
    size_t off = 0;
    auto alloc = [&](size_t bytes) -> void* {
        void* p = (void*)(ws + off);
        off += (bytes + 255) & ~(size_t)255;
        return p;
    };
    unsigned*  mask  = (unsigned*) alloc((size_t)N_NODES * NWORDS * 4);        // 2 MB
    float*     wh1   = (float*)    alloc((size_t)4 * N_NODES * 64 * 4);        // 4 MB
    _Float16*  whT1  = (_Float16*) alloc((size_t)4 * 64 * N_NODES * 2);        // 2 MB
    float*     e1    = (float*)    alloc((size_t)4 * N_NODES * 4);
    float*     f1    = (float*)    alloc((size_t)4 * N_NODES * 4);
    float*     out1  = (float*)    alloc((size_t)N_NODES * 256 * 4);           // 4 MB
    float*     wh2   = (float*)    alloc((size_t)N_NODES * 64 * 4);            // 1 MB
    _Float16*  whT2  = (_Float16*) alloc((size_t)64 * N_NODES * 2);
    float*     e2    = (float*)    alloc((size_t)N_NODES * 4);
    float*     f2    = (float*)    alloc((size_t)N_NODES * 4);

    // 1) adj -> bitmask (the only full read of the 64 MB adjacency)
    pack_adj_kernel<<<(N_NODES * 32) / 256, 256, 0, stream>>>(adj, mask);

    // 2) layer 1 features: wh = h @ Wh + bWh  (4 heads, f32 WMMA)
    gemm_wh_kernel<<<dim3(N_NODES / 16, 4), 32, 0, stream>>>(h, Wh, bWh, wh1, whT1, 256);
    ef_kernel<<<(4 * N_NODES) / 256, 256, 0, stream>>>(wh1, a_src, a_dst, a_b, e1, f1, 4);

    // 3) layer 1 fused masked-softmax attention + ELU -> concat [4096,256]
    attn_kernel<<<dim3(N_NODES / 16, 4), 128, 0, stream>>>(mask, e1, f1, whT1, out1, 256);

    // 4) output layer
    gemm_wh_kernel<<<dim3(N_NODES / 16, 1), 32, 0, stream>>>(out1, Wo, bWo, wh2, whT2, 256);
    ef_kernel<<<N_NODES / 256, 256, 0, stream>>>(wh2, ao_src, ao_dst, ao_b, e2, f2, 1);
    attn_kernel<<<dim3(N_NODES / 16, 1), 128, 0, stream>>>(mask, e2, f2, whT2, (float*)d_out, 64);
}
